// FinalizedQuantizedLinear_52999896433085
// MI455X (gfx1250) — compile-verified, matrix operand order checked by reading, not run on hardware
//
#include <hip/hip_runtime.h>

// ---------------------------------------------------------------------------
// AQLM quantized linear on gfx1250 (MI455X):
//   Phase 1: dequantize codes -> bf16 weight [N=4096, K=4096] into d_ws
//   Phase 2: bf16 WMMA GEMM: out[M=16384, N=4096] = x[M,K] * W^T + bias
// Memory-bound (~0.6 GB HBM traffic -> ~25us at 23.3 TB/s); W_bf16 (33.5 MB)
// stays resident in the 192 MB L2 across the GEMM (out stores are NT so they
// do not thrash L2). Double-buffered LDS; async global->LDS staging for W;
// in-register f32->bf16 conversion for x.
// ---------------------------------------------------------------------------

typedef unsigned short u16;
typedef unsigned int   u32;
typedef __attribute__((ext_vector_type(4)))  int    v4i;
typedef __attribute__((ext_vector_type(8)))  u16    v8u16;
typedef __attribute__((ext_vector_type(4)))  u32    v4u32;
typedef __attribute__((ext_vector_type(2)))  __bf16 v2bf;
typedef __attribute__((ext_vector_type(16))) __bf16 v16bf;
typedef __attribute__((ext_vector_type(8)))  float  v8f;

#define AS1 __attribute__((address_space(1)))
#define AS3 __attribute__((address_space(3)))

#define M_TOT 16384
#define N_TOT 4096
#define K_TOT 4096

#define BM 128
#define BN 128
#define BK 64
#define LDA 72   // padded u16 stride to stagger LDS banks
#define LDB 72
#define NTILES (K_TOT / BK)

#if __has_builtin(__builtin_amdgcn_cvt_pk_bf16_f32)
__device__ __forceinline__ u32 pack_bf16(float a, float b) {
  v2bf r = __builtin_amdgcn_cvt_pk_bf16_f32(a, b);
  return __builtin_bit_cast(u32, r);
}
#else
__device__ __forceinline__ u32 pack_bf16(float a, float b) {
  // compiler-native f32->bf16 (RNE); lets clang pick the hw cvt op
  v2bf r = {(__bf16)a, (__bf16)b};
  return __builtin_bit_cast(u32, r);
}
#endif

struct BFragBits { v8u16 lo, hi; };

__device__ __forceinline__ v16bf load_frag(const u16* p) {
  // 16x32 bf16 fragment, A-style layout: per lane two 16B chunks 32B apart in K
  BFragBits t;
  t.lo = *(const v8u16*)(p);
  t.hi = *(const v8u16*)(p + 16);
  return __builtin_bit_cast(v16bf, t);
}

// ---------------------------------------------------------------------------
// Kernel 1: dequantize additive-quantized weight to bf16.
// ---------------------------------------------------------------------------
__global__ __launch_bounds__(256) void aqlm_dequant_kernel(
    const int* __restrict__ codes,       // [4096, 512, 2]
    const float* __restrict__ codebooks, // [2, 256, 1, 8]
    const float* __restrict__ scales,    // [4096]
    u16* __restrict__ wq)                // [4096, 4096] bf16 bits
{
  __shared__ float cb[2 * 256 * 8];      // 16 KB
  const int t = threadIdx.x;
#pragma unroll
  for (int i = 0; i < 16; ++i) cb[t + i * 256] = codebooks[t + i * 256];
  __syncthreads();

  const long g = (long)blockIdx.x * 256 + t;   // group id: o*512 + ig
  const int  o = (int)(g >> 9);
  const int2 c = *(const int2*)(codes + g * 2);
  const float s = scales[o];

  const float4* e0 = (const float4*)&cb[c.x * 8];
  const float4* e1 = (const float4*)&cb[(256 + c.y) * 8];
  float4 a0 = e0[0], a1 = e0[1];
  float4 b0 = e1[0], b1 = e1[1];

  v4u32 q;
  q[0] = pack_bf16((a0.x + b0.x) * s, (a0.y + b0.y) * s);
  q[1] = pack_bf16((a0.z + b0.z) * s, (a0.w + b0.w) * s);
  q[2] = pack_bf16((a1.x + b1.x) * s, (a1.y + b1.y) * s);
  q[3] = pack_bf16((a1.z + b1.z) * s, (a1.w + b1.w) * s);
  *(v8u16*)(wq + g * 8) = __builtin_bit_cast(v8u16, q);
}

// ---------------------------------------------------------------------------
// Staging helpers
// ---------------------------------------------------------------------------
__device__ __forceinline__ void stage_x(const float* __restrict__ x,
                                        u16* __restrict__ dstbase,
                                        int m_base, int k0, int srow, int scol) {
  const float4* src = (const float4*)(x + (size_t)(m_base + srow) * K_TOT + k0 + scol);
  v8u16* dst = (v8u16*)(dstbase + srow * LDA + scol);
#pragma unroll
  for (int i = 0; i < 4; ++i) {
    float4 f0 = src[2 * i];
    float4 f1 = src[2 * i + 1];
    v4u32 q;
    q[0] = pack_bf16(f0.x, f0.y);
    q[1] = pack_bf16(f0.z, f0.w);
    q[2] = pack_bf16(f1.x, f1.y);
    q[3] = pack_bf16(f1.z, f1.w);
    dst[i] = __builtin_bit_cast(v8u16, q);
  }
}

__device__ __forceinline__ void stage_w(const u16* __restrict__ wq,
                                        u16* __restrict__ dstbase,
                                        int n_base, int k0, int srow, int scol) {
  const u16* src = wq + (size_t)(n_base + srow) * K_TOT + k0 + scol;
  u16* dst = dstbase + srow * LDB + scol;
#if __has_builtin(__builtin_amdgcn_global_load_async_to_lds_b128)
#pragma unroll
  for (int i = 0; i < 4; ++i)
    __builtin_amdgcn_global_load_async_to_lds_b128(
        (AS1 v4i*)(src + 8 * i), (AS3 v4i*)(dst + 8 * i), 0, 0);
#else
#pragma unroll
  for (int i = 0; i < 4; ++i)
    ((v8u16*)dst)[i] = ((const v8u16*)src)[i];
#endif
}

__device__ __forceinline__ void wait_async_lds() {
#if __has_builtin(__builtin_amdgcn_s_wait_asynccnt)
  __builtin_amdgcn_s_wait_asynccnt(0);
#endif
}

// ---------------------------------------------------------------------------
// Kernel 2: bf16 WMMA GEMM with f32 accumulate + bias; double-buffered LDS.
// 256 threads = 8 waves; 128x128 C tile; wave grid 4(M) x 2(N), each wave
// 32x64 = 2x4 subtiles of v_wmma_f32_16x16x32_bf16.
// ---------------------------------------------------------------------------
__global__ __launch_bounds__(256) void aqlm_gemm_kernel(
    const float* __restrict__ x,    // [16384, 4096] f32
    const u16* __restrict__ wq,     // [4096, 4096] bf16 bits (N-major)
    const float* __restrict__ bias, // [4096]
    float* __restrict__ out)        // [16384, 4096] f32
{
  __shared__ u16 lds_a[2][BM * LDA];   // 2 x 18 KB
  __shared__ u16 lds_b[2][BN * LDB];   // 2 x 18 KB

  const int t    = threadIdx.x;
  const int lane = t & 31;
  const int wave = t >> 5;
  const int half = lane >> 4;
  const int r    = lane & 15;

  const int m_base = blockIdx.y * BM;
  const int n_base = blockIdx.x * BN;
  const int m_off  = (wave & 3) * 32;   // 4 waves along M
  const int n_off  = (wave >> 2) * 64;  // 2 waves along N

  const int srow = t >> 1;        // staging: 2 threads per row
  const int scol = (t & 1) * 32;  // 32 elements each

  v8f acc[2][4];
#pragma unroll
  for (int i = 0; i < 2; ++i)
#pragma unroll
    for (int j = 0; j < 4; ++j) {
      v8f z = {0.f, 0.f, 0.f, 0.f, 0.f, 0.f, 0.f, 0.f};
      acc[i][j] = z;
    }

  // prologue: stage tile 0 into buffer 0
  stage_x(x, lds_a[0], m_base, 0, srow, scol);
  stage_w(wq, lds_b[0], n_base, 0, srow, scol);
  wait_async_lds();
  __syncthreads();

  for (int it = 0; it < NTILES; ++it) {
    const int b = it & 1;

    // stage tile it+1 into the other buffer while computing tile it
    if (it + 1 < NTILES) {
      stage_x(x, lds_a[b ^ 1], m_base, (it + 1) * BK, srow, scol);
      stage_w(wq, lds_b[b ^ 1], n_base, (it + 1) * BK, srow, scol);
      if (it + 2 < NTILES)
        __builtin_prefetch(x + (size_t)(m_base + srow) * K_TOT + (it + 2) * BK + scol, 0, 1);
    }

    // compute tile it from buffer b
    const u16* la = lds_a[b];
    const u16* lb = lds_b[b];
#pragma unroll
    for (int kk = 0; kk < BK; kk += 32) {
      v16bf af[2], bf[4];
#pragma unroll
      for (int i = 0; i < 2; ++i)
        af[i] = load_frag(&la[(m_off + 16 * i + r) * LDA + kk + half * 8]);
#pragma unroll
      for (int j = 0; j < 4; ++j)
        bf[j] = load_frag(&lb[(n_off + 16 * j + r) * LDB + kk + half * 8]);
#pragma unroll
      for (int i = 0; i < 2; ++i)
#pragma unroll
        for (int j = 0; j < 4; ++j)
          acc[i][j] = __builtin_amdgcn_wmma_f32_16x16x32_bf16(
              false, af[i], false, bf[j], (short)0, acc[i][j], false, false);
    }

    // my async copies into b^1 must land before anyone reads them (post-barrier)
    wait_async_lds();
    __syncthreads();
  }

  // epilogue: bias add + non-temporal store (out is write-once, keep L2 for x/W)
  // C layout: VGPR v -> M = v + 8*half, N = r
#pragma unroll
  for (int i = 0; i < 2; ++i) {
    const int mrow = m_base + m_off + 16 * i + 8 * half;
#pragma unroll
    for (int j = 0; j < 4; ++j) {
      const int col = n_base + n_off + 16 * j + r;
      const float bv = bias[col];
      float* po = out + (size_t)mrow * N_TOT + col;
#pragma unroll
      for (int v = 0; v < 8; ++v)
        __builtin_nontemporal_store(acc[i][j][v] + bv, po + (size_t)v * N_TOT);
    }
  }
}

// ---------------------------------------------------------------------------
extern "C" void kernel_launch(void* const* d_in, const int* in_sizes, int n_in,
                              void* d_out, int out_size, void* d_ws, size_t ws_size,
                              hipStream_t stream) {
  const float* x         = (const float*)d_in[0];
  const int*   codes     = (const int*)d_in[1];
  const float* codebooks = (const float*)d_in[2];
  const float* scales    = (const float*)d_in[3];
  const float* bias      = (const float*)d_in[4];
  float* out = (float*)d_out;
  u16*   wq  = (u16*)d_ws;   // 4096*4096*2 = 33.5 MB bf16 weight

  aqlm_dequant_kernel<<<8192, 256, 0, stream>>>(codes, codebooks, scales, wq);

  dim3 grid(N_TOT / BN, M_TOT / BM);   // (32, 128): N fastest -> x stripe L2 reuse
  aqlm_gemm_kernel<<<grid, 256, 0, stream>>>(x, wq, bias, out);
}
